// Attention_87308095193199
// MI455X (gfx1250) — compile-verified, matrix operand order checked by reading, not run on hardware
//
#include <hip/hip_runtime.h>
#include <hip/hip_bf16.h>

typedef __bf16 bf16_t;
typedef __attribute__((ext_vector_type(16))) __bf16 v16bf;
typedef __attribute__((ext_vector_type(8)))  __bf16 v8bf;
typedef __attribute__((ext_vector_type(8)))  float  v8f;
typedef __attribute__((ext_vector_type(4)))  float  v4f;
typedef __attribute__((ext_vector_type(4)))  unsigned int u32x4;
typedef __attribute__((ext_vector_type(4)))  int    i32x4;
typedef __attribute__((ext_vector_type(8)))  int    i32x8;

#define D_MODEL 1024
#define HEAD    128
#define SEQ     4096
#define BATCH   4
#define ROWS    (BATCH * SEQ)   // 16384

#if __has_builtin(__builtin_amdgcn_tensor_load_to_lds) && \
    __has_builtin(__builtin_amdgcn_s_wait_tensorcnt)
#define USE_TDM 1
#else
#define USE_TDM 0
#endif

__device__ __forceinline__ v8f zero8() {
  v8f z;
#pragma unroll
  for (int i = 0; i < 8; ++i) z[i] = 0.0f;
  return z;
}

__device__ __forceinline__ v16bf pack16(v8bf lo, v8bf hi) {
  v16bf r;
#pragma unroll
  for (int j = 0; j < 8; ++j) { r[j] = lo[j]; r[8 + j] = hi[j]; }
  return r;
}

__device__ __forceinline__ v8f wmma_bf16(v16bf a, v16bf b, v8f c) {
  // D = A(16x32 bf16) x B(32x16 bf16) + C(16x16 f32)
  return __builtin_amdgcn_wmma_f32_16x16x32_bf16(false, a, false, b, (short)0, c,
                                                 false, false);
}

#if USE_TDM
// Issue one TDM 2D-tile load: tensor is [tensor_d1][tensor_d0] bf16 row-major
// with row stride stride0 (elements); tile is [tile_d1][tile_d0] starting at
// gaddr (bytes), compacted row-major into LDS at lds_addr (bytes).
__device__ __forceinline__ void tdm_load_2d(unsigned lds_addr,
                                            unsigned long long gaddr,
                                            unsigned tensor_d0,
                                            unsigned tensor_d1,
                                            unsigned stride0,
                                            unsigned tile_d0,
                                            unsigned tile_d1) {
  u32x4 g0;
  g0[0] = 1u;                                            // count=1, user D#
  g0[1] = lds_addr;                                      // lds_addr [63:32]
  g0[2] = (unsigned)(gaddr & 0xFFFFFFFFu);               // global_addr lo
  g0[3] = (unsigned)((gaddr >> 32) & 0x1FFFFFFu) | 0x80000000u;  // hi | type=2
  i32x8 g1;
  g1[0] = 0x00010000;                                    // data_size=1 (2B)
  g1[1] = (int)((tensor_d0 & 0xFFFFu) << 16);            // tensor_dim0 lo16
  g1[2] = (int)((tensor_d0 >> 16) | ((tensor_d1 & 0xFFFFu) << 16));
  g1[3] = (int)((tensor_d1 >> 16) | (tile_d0 << 16));    // tile_dim0
  g1[4] = (int)(tile_d1 & 0xFFFFu);                      // tile_dim1, dim2=0
  g1[5] = (int)stride0;                                  // dim0 stride lo32
  g1[6] = 0;
  g1[7] = 0;
  i32x4 z4;
  z4[0] = 0; z4[1] = 0; z4[2] = 0; z4[3] = 0;
  i32x8 z8;
#pragma unroll
  for (int i = 0; i < 8; ++i) z8[i] = 0;
  // This toolchain exposes the 6-arg form:
  // (g0, g1, g2, g3, <extra group>, cpol)
  __builtin_amdgcn_tensor_load_to_lds(g0, g1, z4, z4, z8, 0);
}
#endif

// ---------------------------------------------------------------------------
// Kernel 0: transpose + cast weights: W[k][n] f32 (1024x128) -> Wt[n][k] bf16
// ---------------------------------------------------------------------------
__global__ __launch_bounds__(256) void wcast_kernel(
    const float* __restrict__ Wq, const float* __restrict__ Wk,
    const float* __restrict__ Wv, bf16_t* __restrict__ Wqt,
    bf16_t* __restrict__ Wkt, bf16_t* __restrict__ Wvt) {
  int idx = blockIdx.x * blockDim.x + threadIdx.x;  // 0 .. 1024*128-1
  int k = idx >> 7;     // 0..1023
  int n = idx & 127;    // 0..127
  int o = n * D_MODEL + k;
  Wqt[o] = (bf16_t)Wq[idx];
  Wkt[o] = (bf16_t)Wk[idx];
  Wvt[o] = (bf16_t)Wv[idx];
}

// ---------------------------------------------------------------------------
// Kernel 1: QKV projection. Block = 8 waves, each wave owns a 16-col slice of
// the 128 head dims for one 16-row tile. bf16 WMMA, f32 accumulate.
// Q gets 1/sqrt(HEAD) folded in. V is stored transposed: Vt[b][h][s].
// ---------------------------------------------------------------------------
__global__ __launch_bounds__(256) void qkv_kernel(
    const float* __restrict__ x, const bf16_t* __restrict__ Wqt,
    const bf16_t* __restrict__ Wkt, const bf16_t* __restrict__ Wvt,
    bf16_t* __restrict__ Qo, bf16_t* __restrict__ Ko,
    bf16_t* __restrict__ Vt) {
  const int lane = threadIdx.x & 31;
  const int wave = threadIdx.x >> 5;   // 0..7 -> head-dim slice
  const int l    = lane & 15;
  const int half = lane >> 4;
  const int m0   = blockIdx.x * 16;    // tile over 16384 rows
  const int n    = wave * 16 + l;      // output column (head dim)

  const float*  xrow = x + (size_t)(m0 + l) * D_MODEL;  // A-frag row = lane%16
  const bf16_t* wq   = Wqt + (size_t)n * D_MODEL;
  const bf16_t* wk   = Wkt + (size_t)n * D_MODEL;
  const bf16_t* wv   = Wvt + (size_t)n * D_MODEL;

  v8f cq = zero8(), ck = zero8(), cv = zero8();

  for (int k0 = 0; k0 < D_MODEL; k0 += 32) {
    const int ka = k0 + half * 8;
    // A fragment: x row, K = {ka..ka+7, ka+16..ka+23}, cast f32->bf16
    v4f a0 = *(const v4f*)(xrow + ka);
    v4f a1 = *(const v4f*)(xrow + ka + 4);
    v4f a2 = *(const v4f*)(xrow + ka + 16);
    v4f a3 = *(const v4f*)(xrow + ka + 20);
    v16bf a;
#pragma unroll
    for (int j = 0; j < 4; ++j) {
      a[j]      = (bf16_t)a0[j];
      a[4 + j]  = (bf16_t)a1[j];
      a[8 + j]  = (bf16_t)a2[j];
      a[12 + j] = (bf16_t)a3[j];
    }
    // B fragments from transposed weights: contiguous bf16
    v16bf bq = pack16(*(const v8bf*)(wq + ka), *(const v8bf*)(wq + ka + 16));
    v16bf bk = pack16(*(const v8bf*)(wk + ka), *(const v8bf*)(wk + ka + 16));
    v16bf bv = pack16(*(const v8bf*)(wv + ka), *(const v8bf*)(wv + ka + 16));

    cq = wmma_bf16(a, bq, cq);
    ck = wmma_bf16(a, bk, ck);
    cv = wmma_bf16(a, bv, cv);
  }

  const float qscale = 0.08838834764831845f;  // 1/sqrt(128) folded into Q
#pragma unroll
  for (int r = 0; r < 8; ++r) {
    int m = m0 + r + 8 * half;   // C layout: M = vgpr + 8*(lane/16)
    Qo[(size_t)m * HEAD + n] = (bf16_t)(cq[r] * qscale);
    Ko[(size_t)m * HEAD + n] = (bf16_t)ck[r];
    int b = m >> 12;             // /4096 (tiles never straddle batches)
    int s = m & (SEQ - 1);
    Vt[(size_t)b * HEAD * SEQ + (size_t)n * SEQ + s] = (bf16_t)cv[r];
  }
}

// ---------------------------------------------------------------------------
// Kernel 2: flash attention. 8 waves/block, one 16-query tile per wave, all
// waves share K/V key-block tiles staged in LDS via TDM (double-buffered).
// Computes S^T = K.Q^T so queries live one-per-lane and the S^T C-layout
// feeds the P^T B-frag for O^T = V^T . P^T with no cross-lane movement.
// ---------------------------------------------------------------------------
__global__ __launch_bounds__(256) void attn_kernel(
    const bf16_t* __restrict__ Q, const bf16_t* __restrict__ K,
    const bf16_t* __restrict__ Vt, float* __restrict__ out) {
  __shared__ bf16_t sK[2][32][HEAD];   // key-block rows, head-major (8KB x2)
  __shared__ bf16_t sV[2][HEAD][32];   // V^T: head rows, key-major (8KB x2)

  const int lane = threadIdx.x & 31;
  const int wave = threadIdx.x >> 5;
  const int l    = lane & 15;
  const int half = lane >> 4;
  const int b    = blockIdx.y;
  const int q0   = (blockIdx.x * 8 + wave) * 16;

  const bf16_t* Kb = K + (size_t)b * SEQ * HEAD;
  const bf16_t* Vb = Vt + (size_t)b * HEAD * SEQ;

  // ---- tile staging (TDM when available, cooperative loads otherwise) ----
#if USE_TDM
  auto stage = [&](int pbuf, int kb) {
    if (wave == 0) {
      unsigned ldsK = (unsigned)(size_t)&sK[pbuf][0][0];
      unsigned ldsV = (unsigned)(size_t)&sV[pbuf][0][0];
      unsigned long long gK =
          (unsigned long long)(size_t)(Kb + (size_t)kb * HEAD);
      unsigned long long gV = (unsigned long long)(size_t)(Vb + kb);
      // K tile: tensor 4096x128, tile 32 rows x 128 cols
      tdm_load_2d(ldsK, gK, HEAD, SEQ, HEAD, HEAD, 32);
      // V^T tile: tensor 128x4096, tile 128 rows x 32 cols
      tdm_load_2d(ldsV, gV, SEQ, HEAD, SEQ, 32, HEAD);
    }
  };
#else
  auto stage = [&](int pbuf, int kb) {
    for (int c = threadIdx.x; c < 512; c += 256) {
      {  // K tile: 32 rows x 16 chunks of 16B
        int row = c >> 4, col = (c & 15) * 8;
        *(v8bf*)&sK[pbuf][row][col] =
            *(const v8bf*)(Kb + (size_t)(kb + row) * HEAD + col);
      }
      {  // V^T tile: 128 rows x 4 chunks of 16B
        int row = c >> 2, col = (c & 3) * 8;
        *(v8bf*)&sV[pbuf][row][col] =
            *(const v8bf*)(Vb + (size_t)row * SEQ + kb + col);
      }
    }
  };
#endif

  // Q^T B-fragments (4 chunks of K=32 over the 128 head dims), loaded once.
  const bf16_t* qrow = Q + ((size_t)b * SEQ + q0 + l) * HEAD;
  v16bf qf[4];
#pragma unroll
  for (int c = 0; c < 4; ++c) {
    int ko = c * 32 + half * 8;
    qf[c] = pack16(*(const v8bf*)(qrow + ko), *(const v8bf*)(qrow + ko + 16));
  }

  v8f acc[8];
#pragma unroll
  for (int h = 0; h < 8; ++h) acc[h] = zero8();
  float m_i = -INFINITY;
  float l_i = 0.0f;

  stage(0, 0);  // prologue: tile 0 -> buffer 0

  for (int it = 0; it < SEQ / 32; ++it) {
    const int p  = it & 1;
    const int kb = it * 32;
#if USE_TDM
    if (wave == 0) __builtin_amdgcn_s_wait_tensorcnt(0);
#endif
    __syncthreads();  // tile `it` resident; buffer p^1 free for prefetch
    if (it + 1 < SEQ / 32) stage(p ^ 1, kb + 32);

    // --- scores: S^T two 16-key subtiles, accumulate over 4 head chunks ---
    v8f st0 = zero8(), st1 = zero8();
    const bf16_t* k0row = &sK[p][l][0];       // keys kb..kb+15
    const bf16_t* k1row = &sK[p][16 + l][0];  // keys kb+16..kb+31
#pragma unroll
    for (int c = 0; c < 4; ++c) {
      int ko = c * 32 + half * 8;
      v16bf af0 =
          pack16(*(const v8bf*)(k0row + ko), *(const v8bf*)(k0row + ko + 16));
      v16bf af1 =
          pack16(*(const v8bf*)(k1row + ko), *(const v8bf*)(k1row + ko + 16));
      st0 = wmma_bf16(af0, qf[c], st0);
      st1 = wmma_bf16(af1, qf[c], st1);
    }

    // --- online softmax over this query's 32 keys (split across halves) ---
    float tmax = st0[0];
#pragma unroll
    for (int r = 0; r < 8; ++r) {
      tmax = fmaxf(tmax, st0[r]);
      tmax = fmaxf(tmax, st1[r]);
    }
    tmax = fmaxf(tmax, __shfl_xor(tmax, 16, 32));
    float m_new = fmaxf(m_i, tmax);

    float p0[8], p1[8];
    float psum = 0.0f;
#pragma unroll
    for (int r = 0; r < 8; ++r) {
      p0[r] = __expf(st0[r] - m_new);
      p1[r] = __expf(st1[r] - m_new);
      psum += p0[r] + p1[r];
    }
    psum += __shfl_xor(psum, 16, 32);

    float alpha = __expf(m_i - m_new);
    l_i = l_i * alpha + psum;
    m_i = m_new;

    // P^T B-fragment: C-layout of st0/st1 matches B layout directly.
    v16bf pb;
#pragma unroll
    for (int r = 0; r < 8; ++r) {
      pb[r]     = (bf16_t)p0[r];
      pb[8 + r] = (bf16_t)p1[r];
    }

    // --- O^T = alpha*O^T + V^T . P^T ---
#pragma unroll
    for (int h = 0; h < 8; ++h) {
#pragma unroll
      for (int r = 0; r < 8; ++r) acc[h][r] *= alpha;
      const bf16_t* vrow = &sV[p][h * 16 + l][half * 8];
      v16bf vf = pack16(*(const v8bf*)(vrow), *(const v8bf*)(vrow + 16));
      acc[h] = wmma_bf16(vf, pb, acc[h]);
    }
  }

  // --- normalize and store: O[b][q][head], head = h*16 + r + 8*half ---
  float rinv = 1.0f / l_i;
  float* orow = out + ((size_t)b * SEQ + q0 + l) * HEAD;
#pragma unroll
  for (int h = 0; h < 8; ++h) {
    v4f o0, o1;
#pragma unroll
    for (int r = 0; r < 4; ++r) {
      o0[r] = acc[h][r] * rinv;
      o1[r] = acc[h][4 + r] * rinv;
    }
    *(v4f*)(orow + h * 16 + 8 * half)     = o0;
    *(v4f*)(orow + h * 16 + 8 * half + 4) = o1;
  }
}

// ---------------------------------------------------------------------------
extern "C" void kernel_launch(void* const* d_in, const int* in_sizes, int n_in,
                              void* d_out, int out_size, void* d_ws,
                              size_t ws_size, hipStream_t stream) {
  (void)in_sizes; (void)n_in; (void)out_size; (void)ws_size;
  const float* x  = (const float*)d_in[0];
  const float* Wq = (const float*)d_in[1];
  const float* Wk = (const float*)d_in[2];
  const float* Wv = (const float*)d_in[3];

  char* ws = (char*)d_ws;
  size_t off = 0;
  bf16_t* Wqt = (bf16_t*)(ws + off); off += (size_t)HEAD * D_MODEL * sizeof(bf16_t);
  bf16_t* Wkt = (bf16_t*)(ws + off); off += (size_t)HEAD * D_MODEL * sizeof(bf16_t);
  bf16_t* Wvt = (bf16_t*)(ws + off); off += (size_t)HEAD * D_MODEL * sizeof(bf16_t);
  bf16_t* Qb  = (bf16_t*)(ws + off); off += (size_t)ROWS * HEAD * sizeof(bf16_t);
  bf16_t* Kb  = (bf16_t*)(ws + off); off += (size_t)ROWS * HEAD * sizeof(bf16_t);
  bf16_t* Vtb = (bf16_t*)(ws + off); off += (size_t)ROWS * HEAD * sizeof(bf16_t);

  wcast_kernel<<<(HEAD * D_MODEL) / 256, 256, 0, stream>>>(Wq, Wk, Wv, Wqt, Wkt, Wvt);
  qkv_kernel<<<ROWS / 16, 256, 0, stream>>>(x, Wqt, Wkt, Wvt, Qb, Kb, Vtb);
  attn_kernel<<<dim3(SEQ / 128, BATCH), 256, 0, stream>>>(Qb, Kb, Vtb, (float*)d_out);
}
